// AdaptiveRouter_46686294507693
// MI455X (gfx1250) — matrix operand order repaired and checked by reading, not compiled, not run on hardware
//
#include <hip/hip_runtime.h>
#include <hip/hip_bf16.h>
#include <math.h>

// ---------------------------------------------------------------------------
// AdaptiveRouter forward for MI455X (gfx1250, wave32, WMMA).
//
// Roofline: ~105 GFLOP of GEMM vs ~165MB mandatory HBM traffic (~7us at
// 23.3 TB/s) => compute bound. Weights (<6.5MB as bf16) stay L2-resident
// (192MB L2). Strategy: fp32 weights -> bf16 (pre-transposed to [N][K]) once,
// activations converted to bf16 on the fly, all linears on
// v_wmma_f32_16x16x32_bf16 with fp32 accumulation; LN after every linear
// absorbs bf16 rounding. Each 8-wave block owns 16 rows x full N so the
// LayerNorm fuses into the GEMM epilogue. Tile staging uses the CDNA5 async
// global->LDS DMA path (GLOBAL_LOAD_ASYNC_TO_LDS_B64 + s_wait_asynccnt) when
// the toolchain exposes it, avoiding the VGPR round-trip.
// ---------------------------------------------------------------------------

typedef __attribute__((ext_vector_type(16))) __bf16 v16bf;
typedef __attribute__((ext_vector_type(8)))  float  v8f;
typedef __attribute__((ext_vector_type(2)))  int    v2i;

#define TOKENS 16384   // B * S = 4 * 4096

#if defined(__HIP_DEVICE_COMPILE__) && __has_builtin(__builtin_amdgcn_global_load_async_to_lds_b64)
#define HAVE_ASYNC_LDS 1
#else
#define HAVE_ASYNC_LDS 0
#endif

#define AS1 __attribute__((address_space(1)))
#define AS3 __attribute__((address_space(3)))

__device__ __forceinline__ void wait_async_lds() {
#if HAVE_ASYNC_LDS
#if __has_builtin(__builtin_amdgcn_s_wait_asynccnt)
    __builtin_amdgcn_s_wait_asynccnt(0);
#else
    asm volatile("s_wait_asynccnt 0x0" ::: "memory");
#endif
#endif
}

#if HAVE_ASYNC_LDS
// per-lane async copy of 8 bytes: global -> LDS (tracked by ASYNCcnt)
__device__ __forceinline__ void async_copy_b64(const __bf16* gsrc, __bf16* ldst) {
    __builtin_amdgcn_global_load_async_to_lds_b64(
        (AS1 v2i*)(AS1 void*)const_cast<__bf16*>(gsrc),
        (AS3 v2i*)(AS3 void*)ldst, 0, 0);
}
#endif

// ---------------------------------------------------------------------------
// fp32 [K,N] -> bf16 transposed [N,K] (weights; one-time, L2-resident after)
// ---------------------------------------------------------------------------
__global__ __launch_bounds__(256) void f2bf_t_kernel(const float* __restrict__ src,
                                                     __bf16* __restrict__ dst,
                                                     int K, int N) {
    int i = blockIdx.x * 256 + threadIdx.x;
    if (i < K * N) {
        int k = i / N, n = i - k * N;
        dst[(size_t)n * K + k] = (__bf16)src[i];
    }
}

// ---------------------------------------------------------------------------
// Fused GEMM (bf16 WMMA, fp32 accum) + bias + LayerNorm + erf-GELU.
//   A:[TOKENS,K] bf16 (or concat(x,cot) fp32 when CONCAT), Wt:[N,K] bf16,
//   Out:[TOKENS,N] bf16.  Block = 256 threads (8 waves), owns 16 rows x N.
// ---------------------------------------------------------------------------
template<int K, int N, bool CONCAT>
__global__ __launch_bounds__(256) void gemm_ln_gelu(
    const __bf16* __restrict__ A,
    const float*  __restrict__ X,      // concat source 0 [TOKENS,2048]
    const float*  __restrict__ Cot,    // concat source 1 [TOKENS,256]
    const __bf16* __restrict__ Wt,     // [N,K] (transposed)
    const float*  __restrict__ bias,   // [N]
    const float*  __restrict__ gamma,  // [N]
    const float*  __restrict__ beta,   // [N]
    __bf16* __restrict__ Out)          // [TOKENS,N]
{
    static_assert(N % 16 == 0 && K % 32 == 0, "tile shape");
    constexpr int LDT    = 36;            // LDS stride (halves): 72B rows, 8B
                                          // aligned for async b64, 18-dword
                                          // stride => conflict-free frag loads
    constexpr int NTILES = N / 16;
    constexpr int NTW    = (NTILES + 7) / 8;  // 16x16 N-tiles per wave

    __shared__ __bf16 sA[16 * LDT];
    union TileU { __bf16 bt[N * LDT]; float o[16 * N]; };
    __shared__ TileU sT;                  // B^T staging, reused as fp32 out tile
    __shared__ float sRed[512];
    __shared__ float sStat[32];           // [0..15]=mean, [16..31]=rstd

    const int tid  = threadIdx.x;
    const int lane = tid & 31;
    const int wave = __builtin_amdgcn_readfirstlane(tid >> 5) & 7; // uniform, 0..7
    const int row0 = blockIdx.x * 16;
    const int hi   = (lane >> 4) & 1;     // lane half-wave

    v8f acc[NTW];
    #pragma unroll
    for (int j = 0; j < NTW; ++j)
        #pragma unroll
        for (int e = 0; e < 8; ++e) acc[j][e] = 0.0f;

    for (int k0 = 0; k0 < K; k0 += 32) {
        // ---- stage A tile [16 x 32 halves] ----
        if (CONCAT) {
            // fp32 concat(x,cot) -> bf16 conversion must go through VGPRs
            for (int e = tid; e < 16 * 32; e += 256) {
                int m = e >> 5, k = e & 31;
                int col = k0 + k;
                float v = (col < 2048) ? X[(size_t)(row0 + m) * 2048 + col]
                                       : Cot[(size_t)(row0 + m) * 256 + (col - 2048)];
                sA[m * LDT + k] = (__bf16)v;
            }
        } else {
#if HAVE_ASYNC_LDS
            if (tid < 128) {              // 16 rows x 8 segs of 8B
                int m = tid >> 3, seg = tid & 7;
                async_copy_b64(A + (size_t)(row0 + m) * K + k0 + seg * 4,
                               &sA[m * LDT + seg * 4]);
            }
#else
            for (int e = tid; e < 16 * 32; e += 256) {
                int m = e >> 5, k = e & 31;
                sA[m * LDT + k] = A[(size_t)(row0 + m) * K + (k0 + k)];
            }
#endif
        }
        // ---- stage B^T tile [N x 32 halves] from transposed weights ----
#if HAVE_ASYNC_LDS
        for (int e = tid; e < N * 8; e += 256) {   // N rows x 8 segs of 8B
            int n = e >> 3, seg = e & 7;
            async_copy_b64(Wt + (size_t)n * K + k0 + seg * 4,
                           &sT.bt[n * LDT + seg * 4]);
        }
#else
        for (int e = tid; e < N * 32; e += 256) {
            int n = e >> 5, k = e & 31;
            sT.bt[n * LDT + k] = Wt[(size_t)n * K + k0 + k];
        }
#endif
        // prefetch next weight k-slab toward L2
        if (k0 + 32 < K)
            __builtin_prefetch(Wt + (size_t)(tid & (N - 1)) * K + k0 + 32, 0, 1);
        wait_async_lds();
        __syncthreads();

        // ---- A fragment (ISA 7.12.2 bf16 A layout): m=lane&15,
        //      lanes0-15: K{0..7,16..23}; lanes16-31: K{8..15,24..31} ----
        const __bf16* ar = &sA[(lane & 15) * LDT + hi * 8];
        v16bf af;
        #pragma unroll
        for (int e = 0; e < 8; ++e) { af[e] = ar[e]; af[e + 8] = ar[e + 16]; }

        #pragma unroll
        for (int j = 0; j < NTW; ++j) {
            const int tile = wave * NTW + j;      // provably < NTILES unless N=64
            if (tile < NTILES) {
                // B fragment: n=lane&15, lanes0-15: K{0..7,16..23}; lanes16-31: K{8..15,24..31}
                const __bf16* br = &sT.bt[(tile * 16 + (lane & 15)) * LDT + hi * 8];
                v16bf bfr;
                #pragma unroll
                for (int e = 0; e < 8; ++e) { bfr[e] = br[e]; bfr[e + 8] = br[e + 16]; }
                acc[j] = __builtin_amdgcn_wmma_f32_16x16x32_bf16(
                    false, af, false, bfr, (short)0, acc[j], false, false);
            }
        }
        __syncthreads();
    }

    // ---- epilogue: accum + bias -> LDS fp32 tile (C/D layout: n=lane&15,
    //      vgpr r -> m = r + 8*(lane>=16)) ----
    #pragma unroll
    for (int j = 0; j < NTW; ++j) {
        const int tile = wave * NTW + j;
        if (tile < NTILES) {
            int n = tile * 16 + (lane & 15);
            float bn = bias[n];
            #pragma unroll
            for (int r = 0; r < 8; ++r)
                sT.o[(r + hi * 8) * N + n] = acc[j][r] + bn;
        }
    }
    __syncthreads();

    // ---- LayerNorm: 16 threads per row reduce N elements ----
    {
        int r = tid >> 4, s = tid & 15;
        float sum = 0.f, sq = 0.f;
        for (int n = s; n < N; n += 16) {
            float v = sT.o[r * N + n];
            sum += v; sq += v * v;
        }
        sRed[r * 16 + s] = sum;
        sRed[256 + r * 16 + s] = sq;
    }
    __syncthreads();
    if (tid < 16) {
        float sum = 0.f, sq = 0.f;
        for (int s = 0; s < 16; ++s) { sum += sRed[tid * 16 + s]; sq += sRed[256 + tid * 16 + s]; }
        float m  = sum / (float)N;
        float vr = sq / (float)N - m * m;
        sStat[tid]      = m;
        sStat[16 + tid] = rsqrtf(vr + 1e-5f);
    }
    __syncthreads();

    // ---- normalize + erf-GELU + bf16 store ----
    for (int e = tid; e < 16 * N; e += 256) {
        int m = e / N, n = e - m * N;
        float v = (sT.o[e] - sStat[m]) * sStat[16 + m] * gamma[n] + beta[n];
        float g = 0.5f * v * (1.0f + erff(v * 0.70710678118654752f));
        Out[(size_t)(row0 + m) * N + n] = (__bf16)g;
    }
}

// ---------------------------------------------------------------------------
// Head second-linears + softmax/sigmoid/argmax/top-2 epilogue.
// One block (64 threads) per token; <1 GFLOP total so VALU is fine here.
// Output tuple layout (floats, concatenated in reference return order).
// ---------------------------------------------------------------------------
__global__ __launch_bounds__(64) void head_finalize(
    const __bf16* __restrict__ Hd,  const __bf16* __restrict__ Hwd,
    const __bf16* __restrict__ Hpt, const __bf16* __restrict__ He,
    const __bf16* __restrict__ Hc,  const __bf16* __restrict__ Hu,
    const float* __restrict__ dw2,  const float* __restrict__ db2,
    const float* __restrict__ wdw2, const float* __restrict__ wdb2,
    const float* __restrict__ ptw2, const float* __restrict__ ptb2,
    const float* __restrict__ ew2,  const float* __restrict__ eb2,
    const float* __restrict__ cw2,  const float* __restrict__ cb2,
    const float* __restrict__ uw2,  const float* __restrict__ ub2,
    const float* __restrict__ wvals,
    float* __restrict__ out)
{
    const int t = blockIdx.x;
    const int n = threadIdx.x;                    // 0..63
    __shared__ float sd[24], sw[4], sp[3], se[64], scu[2], sMS[2];

    { // expert logits (256 -> 64), all 64 threads
        const __bf16* f = He + (size_t)t * 256;
        float a = eb2[n];
        for (int k = 0; k < 256; ++k) a += (float)f[k] * ew2[k * 64 + n];
        se[n] = a;
    }
    if (n < 24) { const __bf16* f = Hd  + (size_t)t * 128; float a = db2[n];
        for (int k = 0; k < 128; ++k) a += (float)f[k] * dw2[k * 24 + n]; sd[n] = a; }
    if (n < 4)  { const __bf16* f = Hwd + (size_t)t * 128; float a = wdb2[n];
        for (int k = 0; k < 128; ++k) a += (float)f[k] * wdw2[k * 4 + n]; sw[n] = a; }
    if (n < 3)  { const __bf16* f = Hpt + (size_t)t * 128; float a = ptb2[n];
        for (int k = 0; k < 128; ++k) a += (float)f[k] * ptw2[k * 3 + n]; sp[n] = a; }
    if (n < 2)  { const __bf16* f = (n ? Hu : Hc) + (size_t)t * 64;
        const float* w = n ? uw2 : cw2; float a = n ? ub2[0] : cb2[0];
        for (int k = 0; k < 64; ++k) a += (float)f[k] * w[k]; scu[n] = a; }
    __syncthreads();

    if (n == 2) { // expert softmax stats
        float m = se[0];
        for (int k = 1; k < 64; ++k) m = fmaxf(m, se[k]);
        float s = 0.f;
        for (int k = 0; k < 64; ++k) s += expf(se[k] - m);
        sMS[0] = m; sMS[1] = s;
    }
    __syncthreads();

    const size_t T = TOKENS;
    const size_t b_dl = 0,      b_dp = 24 * T,  b_dm = 48 * T,
                 b_wl = 72 * T, b_wp = 76 * T,  b_wi = 80 * T,  b_wm = 81 * T,
                 b_pl = 82 * T, b_pp = 85 * T,  b_el = 88 * T,  b_ep = 152 * T,
                 b_ei = 216 * T, b_ew = 218 * T, b_c = 220 * T, b_u = 221 * T;

    if (n < 24) { // depth
        float l = sd[n], p = 1.f / (1.f + expf(-l));
        out[b_dl + (size_t)t * 24 + n] = l;
        out[b_dp + (size_t)t * 24 + n] = p;
        out[b_dm + (size_t)t * 24 + n] = (p > 0.5f) ? 1.f : 0.f;
    }
    { // expert logits / probs
        float l = se[n];
        out[b_el + (size_t)t * 64 + n] = l;
        out[b_ep + (size_t)t * 64 + n] = expf(l - sMS[0]) / sMS[1];
    }
    if (n == 0) { // width softmax + argmax (first-max semantics)
        float m = sw[0];
        for (int k = 1; k < 4; ++k) m = fmaxf(m, sw[k]);
        float ex[4], s = 0.f;
        for (int k = 0; k < 4; ++k) { ex[k] = expf(sw[k] - m); s += ex[k]; }
        int idx = 0; float best = -3.4e38f;
        for (int k = 0; k < 4; ++k) {
            out[b_wl + (size_t)t * 4 + k] = sw[k];
            out[b_wp + (size_t)t * 4 + k] = ex[k] / s;
            if (sw[k] > best) { best = sw[k]; idx = k; }
        }
        out[b_wi + t] = (float)idx;
        out[b_wm + t] = wvals[idx];
    }
    if (n == 1) { // path softmax
        float m = fmaxf(sp[0], fmaxf(sp[1], sp[2]));
        float e0 = expf(sp[0] - m), e1 = expf(sp[1] - m), e2 = expf(sp[2] - m);
        float s = e0 + e1 + e2;
        out[b_pl + (size_t)t * 3 + 0] = sp[0];
        out[b_pl + (size_t)t * 3 + 1] = sp[1];
        out[b_pl + (size_t)t * 3 + 2] = sp[2];
        out[b_pp + (size_t)t * 3 + 0] = e0 / s;
        out[b_pp + (size_t)t * 3 + 1] = e1 / s;
        out[b_pp + (size_t)t * 3 + 2] = e2 / s;
    }
    if (n == 3) { // top-2 experts (descending, ties -> lower index)
        int i1 = 0; float v1 = -3.4e38f;
        for (int k = 0; k < 64; ++k) if (se[k] > v1) { v1 = se[k]; i1 = k; }
        int i2 = 0; float v2 = -3.4e38f;
        for (int k = 0; k < 64; ++k) if (k != i1 && se[k] > v2) { v2 = se[k]; i2 = k; }
        float p1 = expf(v1 - sMS[0]) / sMS[1];
        float p2 = expf(v2 - sMS[0]) / sMS[1];
        float s12 = p1 + p2;
        out[b_ei + (size_t)t * 2 + 0] = (float)i1;
        out[b_ei + (size_t)t * 2 + 1] = (float)i2;
        out[b_ew + (size_t)t * 2 + 0] = p1 / s12;
        out[b_ew + (size_t)t * 2 + 1] = p2 / s12;
    }
    if (n < 2) { // complexity / uncertainty
        out[(n ? b_u : b_c) + t] = 1.f / (1.f + expf(-scu[n]));
    }
}

// ---------------------------------------------------------------------------
// Launch
// ---------------------------------------------------------------------------
extern "C" void kernel_launch(void* const* d_in, const int* in_sizes, int n_in,
                              void* d_out, int out_size, void* d_ws, size_t ws_size,
                              hipStream_t stream) {
    (void)in_sizes; (void)n_in; (void)out_size; (void)ws_size;

    const float* x    = (const float*)d_in[0];
    const float* cot  = (const float*)d_in[1];
    const float* w1   = (const float*)d_in[2];
    const float* b1   = (const float*)d_in[3];
    const float* g1f  = (const float*)d_in[4];
    const float* be1  = (const float*)d_in[5];
    const float* w2   = (const float*)d_in[6];
    const float* b2   = (const float*)d_in[7];
    const float* g2f  = (const float*)d_in[8];
    const float* be2  = (const float*)d_in[9];
    const float* w3   = (const float*)d_in[10];
    const float* b3   = (const float*)d_in[11];
    const float* g3f  = (const float*)d_in[12];
    const float* be3  = (const float*)d_in[13];
    const float* dw1  = (const float*)d_in[14];
    const float* db1  = (const float*)d_in[15];
    const float* dg   = (const float*)d_in[16];
    const float* dbe  = (const float*)d_in[17];
    const float* dw2  = (const float*)d_in[18];
    const float* db2  = (const float*)d_in[19];
    const float* wdw1 = (const float*)d_in[20];
    const float* wdb1 = (const float*)d_in[21];
    const float* wdg  = (const float*)d_in[22];
    const float* wdbe = (const float*)d_in[23];
    const float* wdw2 = (const float*)d_in[24];
    const float* wdb2 = (const float*)d_in[25];
    const float* ptw1 = (const float*)d_in[26];
    const float* ptb1 = (const float*)d_in[27];
    const float* ptg  = (const float*)d_in[28];
    const float* ptbe = (const float*)d_in[29];
    const float* ptw2 = (const float*)d_in[30];
    const float* ptb2 = (const float*)d_in[31];
    const float* ew1  = (const float*)d_in[32];
    const float* eb1  = (const float*)d_in[33];
    const float* eg   = (const float*)d_in[34];
    const float* ebe  = (const float*)d_in[35];
    const float* ew2  = (const float*)d_in[36];
    const float* eb2  = (const float*)d_in[37];
    const float* cw1  = (const float*)d_in[38];
    const float* cb1  = (const float*)d_in[39];
    const float* cg   = (const float*)d_in[40];
    const float* cbe  = (const float*)d_in[41];
    const float* cw2  = (const float*)d_in[42];
    const float* cb2  = (const float*)d_in[43];
    const float* uw1  = (const float*)d_in[44];
    const float* ub1  = (const float*)d_in[45];
    const float* ug   = (const float*)d_in[46];
    const float* ube  = (const float*)d_in[47];
    const float* uw2  = (const float*)d_in[48];
    const float* ub2  = (const float*)d_in[49];
    const float* wval = (const float*)d_in[50];

    // ---- workspace layout ----
    char* ws = (char*)d_ws;
    size_t off = 0;
    auto take = [&](size_t bytes) -> char* {
        char* p = ws + off;
        off = (off + bytes + 255) & ~(size_t)255;
        return p;
    };
    __bf16* w1b  = (__bf16*)take((size_t)2304 * 1024 * 2);
    __bf16* w2b  = (__bf16*)take((size_t)1024 * 512 * 2);
    __bf16* w3b  = (__bf16*)take((size_t)512 * 256 * 2);
    __bf16* dw1b = (__bf16*)take((size_t)256 * 128 * 2);
    __bf16* wdw1b= (__bf16*)take((size_t)256 * 128 * 2);
    __bf16* ptw1b= (__bf16*)take((size_t)256 * 128 * 2);
    __bf16* ew1b = (__bf16*)take((size_t)256 * 256 * 2);
    __bf16* cw1b = (__bf16*)take((size_t)256 * 64 * 2);
    __bf16* uw1b = (__bf16*)take((size_t)256 * 64 * 2);
    __bf16* h1   = (__bf16*)take((size_t)TOKENS * 1024 * 2);
    __bf16* h2   = (__bf16*)take((size_t)TOKENS * 512 * 2);
    __bf16* feat = (__bf16*)take((size_t)TOKENS * 256 * 2);
    // head hiddens alias h1 (h1 dead after layer-2 GEMM; stream-serialized)
    __bf16* hd  = h1;
    __bf16* hwd = h1 + (size_t)TOKENS * 128;
    __bf16* hpt = h1 + (size_t)TOKENS * 256;
    __bf16* he  = h1 + (size_t)TOKENS * 384;
    __bf16* hc  = h1 + (size_t)TOKENS * 640;
    __bf16* hu  = h1 + (size_t)TOKENS * 704;

    // ---- weight conversions to transposed bf16 [N,K] ----
    auto cvt = [&](const float* s, __bf16* d, int K, int N) {
        int ne = K * N;
        f2bf_t_kernel<<<(ne + 255) / 256, 256, 0, stream>>>(s, d, K, N);
    };
    cvt(w1,   w1b,   2304, 1024);
    cvt(w2,   w2b,   1024, 512);
    cvt(w3,   w3b,   512,  256);
    cvt(dw1,  dw1b,  256,  128);
    cvt(wdw1, wdw1b, 256,  128);
    cvt(ptw1, ptw1b, 256,  128);
    cvt(ew1,  ew1b,  256,  256);
    cvt(cw1,  cw1b,  256,  64);
    cvt(uw1,  uw1b,  256,  64);

    const dim3 grid(TOKENS / 16), blk(256);
    // encoder
    gemm_ln_gelu<2304, 1024, true ><<<grid, blk, 0, stream>>>(nullptr, x, cot, w1b, b1, g1f, be1, h1);
    gemm_ln_gelu<1024, 512,  false><<<grid, blk, 0, stream>>>(h1,   nullptr, nullptr, w2b, b2, g2f, be2, h2);
    gemm_ln_gelu<512,  256,  false><<<grid, blk, 0, stream>>>(h2,   nullptr, nullptr, w3b, b3, g3f, be3, feat);
    // head hidden layers
    gemm_ln_gelu<256, 128, false><<<grid, blk, 0, stream>>>(feat, nullptr, nullptr, dw1b,  db1,  dg,  dbe,  hd);
    gemm_ln_gelu<256, 128, false><<<grid, blk, 0, stream>>>(feat, nullptr, nullptr, wdw1b, wdb1, wdg, wdbe, hwd);
    gemm_ln_gelu<256, 128, false><<<grid, blk, 0, stream>>>(feat, nullptr, nullptr, ptw1b, ptb1, ptg, ptbe, hpt);
    gemm_ln_gelu<256, 256, false><<<grid, blk, 0, stream>>>(feat, nullptr, nullptr, ew1b,  eb1,  eg,  ebe,  he);
    gemm_ln_gelu<256, 64,  false><<<grid, blk, 0, stream>>>(feat, nullptr, nullptr, cw1b,  cb1,  cg,  cbe,  hc);
    gemm_ln_gelu<256, 64,  false><<<grid, blk, 0, stream>>>(feat, nullptr, nullptr, uw1b,  ub1,  ug,  ube,  hu);
    // heads + routing epilogue
    head_finalize<<<TOKENS, 64, 0, stream>>>(hd, hwd, hpt, he, hc, hu,
                                             dw2, db2, wdw2, wdb2, ptw2, ptb2,
                                             ew2, eb2, cw2, cb2, uw2, ub2,
                                             wval, (float*)d_out);
}
